// Pol_Net_45243185496396
// MI455X (gfx1250) — matrix operand order repaired and checked by reading, not compiled, not run on hardware
//
#include <hip/hip_runtime.h>

// ---------------------------------------------------------------------------
// MI455X (gfx1250) GNN forward: bf16 WMMA GEMMs + fused gather/concat,
// f32 atomic segment-mean, softmax head.
// Segment boundaries are multiples of 32 -> K loop split into scalar-bounded
// sub-loops (no divergent segment selection in the hot loop).
// ---------------------------------------------------------------------------

#define NN 50000
#define NE 800000

typedef __attribute__((ext_vector_type(16))) __bf16 v16bf;
typedef __attribute__((ext_vector_type(8)))  float  v8f;

struct __attribute__((aligned(16))) U4 { unsigned x, y, z, w; };
union Frag16 { v16bf v; U4 u[2]; };

__device__ __forceinline__ unsigned short f2bf(float f) {
  unsigned u = __float_as_uint(f);
  u += 0x7FFFu + ((u >> 16) & 1u);          // round-to-nearest-even
  return (unsigned short)(u >> 16);
}
__device__ __forceinline__ float bf2f(unsigned short h) {
  return __uint_as_float(((unsigned)h) << 16);
}

// -------------------- convert f32 [M,Kin] -> bf16 [M,Kpad] (zero pad) ------
__global__ void k_convert_pad(const float* __restrict__ src,
                              unsigned short* __restrict__ dst,
                              int M, int Kin, int Kpad) {
  int t = blockIdx.x * 256 + threadIdx.x;
  if (t >= M * Kpad) return;
  int row = t / Kpad, k = t - row * Kpad;
  float v = (k < Kin) ? src[(size_t)row * Kin + k] : 0.0f;
  dst[t] = f2bf(v);
}

__global__ void k_zero(float* __restrict__ p, int n) {
  int t = blockIdx.x * 256 + threadIdx.x;
  if (t < n) p[t] = 0.0f;
}

// -------------------- scatter-add of e (f32) over senders/receivers -------
__global__ void k_scatter(const float* __restrict__ eF,
                          const int* __restrict__ snd, const int* __restrict__ rcv,
                          float* __restrict__ sum_s, float* __restrict__ sum_r,
                          float* __restrict__ cnt_s, float* __restrict__ cnt_r,
                          int E) {
  int t = blockIdx.x * 256 + threadIdx.x;   // E*16 threads, 4 feats each
  if (t >= E * 16) return;
  int edge = t >> 4;
  int q = (t & 15) * 4;
  int s = snd[edge], r = rcv[edge];
  const float* ep = eF + (size_t)edge * 64 + q;
#pragma unroll
  for (int j = 0; j < 4; ++j) {
    float v = ep[j];
    atomicAdd(&sum_s[(size_t)s * 64 + q + j], v);
    atomicAdd(&sum_r[(size_t)r * 64 + q + j], v);
  }
  if ((t & 15) == 0) {
    atomicAdd(&cnt_s[s], 1.0f);
    atomicAdd(&cnt_r[r], 1.0f);
  }
}

// -------------------- mean + concat [agg_s|agg_r] -> bf16 [N,128] ---------
__global__ void k_mean(const float* __restrict__ sum_s, const float* __restrict__ sum_r,
                       const float* __restrict__ cnt_s, const float* __restrict__ cnt_r,
                       unsigned short* __restrict__ agg, int N) {
  int t = blockIdx.x * 256 + threadIdx.x;
  if (t >= N * 128) return;
  int i = t >> 7, j = t & 127;
  float v;
  if (j < 64) v = sum_s[(size_t)i * 64 + j] / fmaxf(cnt_s[i], 1.0f);
  else        v = sum_r[(size_t)i * 64 + (j - 64)] / fmaxf(cnt_r[i], 1.0f);
  agg[t] = f2bf(v);
}

// -------------------- generic WMMA GEMM ------------------------------------
// A is logically [M, Kpad] bf16 built from up to 3 segments (boundaries B1,B2
// are multiples of 32):
//   k <  B1 : A0[(I0?I0[m]:m) * S0 + k]
//   k <  B2 : A1[(I1?I1[m]:m) * S1 + k-B1]
//   else    : A2[(I2?I2[m]:m) * S2 + k-B2]
// W is f32 [Kreal, Nout] row-major (rows >= Kreal treated as zero).
// Each wave: 16x64 tile (4 x v_wmma_f32_16x16x32_bf16 per K step).
__global__ void __launch_bounds__(256)
k_gemm_wmma(const unsigned short* __restrict__ A0, const int* __restrict__ I0, int S0, int B1,
            const unsigned short* __restrict__ A1, const int* __restrict__ I1, int S1, int B2,
            const unsigned short* __restrict__ A2, const int* __restrict__ I2, int S2,
            const float* __restrict__ W, const float* __restrict__ bias,
            int Kpad, int Kreal, int Nout,
            float* __restrict__ outF, unsigned short* __restrict__ outB,
            int Mtiles, int act) {
  extern __shared__ unsigned ldsW[];
  const int tid = threadIdx.x;
  const int colbase = blockIdx.y * 64;
  const int nkb = Kpad >> 5;

  // Stage weights pre-swizzled into B-fragment layout:
  // dword index = kb*1024 + nt*256 + lane*8 + r
  // lane: n = colbase + nt*16 + (lane&15); K-half = (lane>>4)*16; K = ..+2r
  const int total = nkb << 10;
  for (int t = tid; t < total; t += 256) {
    int r = t & 7, lane = (t >> 3) & 31, nt = (t >> 8) & 3, kb = t >> 10;
    int n = colbase + nt * 16 + (lane & 15);
    int k = kb * 32 + ((lane >> 4) << 4) + (r << 1);
    float w0 = (k     < Kreal) ? W[(size_t)k * Nout + n]       : 0.0f;
    float w1 = (k + 1 < Kreal) ? W[(size_t)(k + 1) * Nout + n] : 0.0f;
    ldsW[t] = (unsigned)f2bf(w0) | ((unsigned)f2bf(w1) << 16);
  }
  __syncthreads();

  const int wave = tid >> 5, lane = tid & 31;
  const int tileM = blockIdx.x * 8 + wave;
  if (tileM >= Mtiles) return;
  const int rowbase = tileM << 4;
  const int mrow = rowbase + (lane & 15);
  const int half8 = (lane >> 4) << 4;     // A-fragment K split per lane half

  v8f acc[4];
#pragma unroll
  for (int nt = 0; nt < 4; ++nt)
#pragma unroll
    for (int v = 0; v < 8; ++v) acc[nt][v] = 0.0f;

  // One K-span with a fixed per-lane row pointer; Pseg indexed by GLOBAL k0
  // (caller pre-subtracts the segment start). Scalar loop bounds only.
  auto span = [&](const unsigned short* Pseg, int kbeg, int kend) {
    for (int k0 = kbeg; k0 < kend; k0 += 32) {
      Frag16 af;
      af.u[0] = *(const U4*)(Pseg + k0 + half8);        // VGPR 0-3: K=k..k+7
      af.u[1] = *(const U4*)(Pseg + k0 + 8 + half8);    // VGPR 4-7
      const int lwb = ((k0 >> 5) << 10) + (lane << 3);
#pragma unroll
      for (int nt = 0; nt < 4; ++nt) {
        Frag16 bfr;
        const U4* bp = (const U4*)&ldsW[lwb + (nt << 8)];
        bfr.u[0] = bp[0];
        bfr.u[1] = bp[1];
        acc[nt] = __builtin_amdgcn_wmma_f32_16x16x32_bf16(
            false, af.v, false, bfr.v, (short)0, acc[nt], false, false);
      }
    }
  };

  const int b1c = (B1 < Kpad) ? B1 : Kpad;
  const int b2c = (B2 < Kpad) ? B2 : Kpad;
  if (0 < b1c) {
    const int g0 = I0 ? I0[mrow] : mrow;
    span(A0 + (size_t)g0 * S0, 0, b1c);
  }
  if (B1 < b2c) {
    const int g1 = I1 ? I1[mrow] : mrow;
    span(A1 + (size_t)g1 * S1 - B1, B1, b2c);
  }
  if (B2 < Kpad) {
    const int g2 = I2 ? I2[mrow] : mrow;
    span(A2 + (size_t)g2 * S2 - B2, B2, Kpad);
  }

#pragma unroll
  for (int nt = 0; nt < 4; ++nt) {
    int col = colbase + nt * 16 + (lane & 15);
    float bc = bias[col];
#pragma unroll
    for (int v = 0; v < 8; ++v) {
      int row = rowbase + ((lane < 16) ? v : v + 8);
      float x = acc[nt][v] + bc;
      if (act) x = (x > 0.0f) ? x : 0.01f * x;   // jax leaky_relu slope 0.01
      size_t o = (size_t)row * Nout + col;
      if (outF) outF[o] = x;
      if (outB) outB[o] = f2bf(x);
    }
  }
}

// -------------------- tiny head: out[i,c] = h[i,:] @ W2[:,c] + b2 ----------
__global__ void k_head(const unsigned short* __restrict__ h,
                       const float* __restrict__ W, const float* __restrict__ b,
                       float* __restrict__ out, int N, int C) {
  int t = blockIdx.x * 256 + threadIdx.x;
  if (t >= N * C) return;
  int i = t / C, c = t - i * C;
  const unsigned short* hp = h + (size_t)i * 128;
  float s = 0.0f;
#pragma unroll 8
  for (int k = 0; k < 128; ++k) s += bf2f(hp[k]) * W[k * C + c];
  out[t] = s + b[c];
}

// -------------------- softmax over all N rows (single graph) ---------------
__global__ void k_softmax(const float* __restrict__ logits,
                          float* __restrict__ prob, int N) {
  __shared__ float red[1024];
  int tid = threadIdx.x;
  float m = -3.4e38f;
  for (int i = tid; i < N; i += 1024) m = fmaxf(m, logits[i]);
  red[tid] = m; __syncthreads();
  for (int s = 512; s > 0; s >>= 1) {
    if (tid < s) red[tid] = fmaxf(red[tid], red[tid + s]);
    __syncthreads();
  }
  float mx = red[0]; __syncthreads();
  float sum = 0.0f;
  for (int i = tid; i < N; i += 1024) sum += __expf(logits[i] - mx);
  red[tid] = sum; __syncthreads();
  for (int s = 512; s > 0; s >>= 1) {
    if (tid < s) red[tid] += red[tid + s];
    __syncthreads();
  }
  float inv = 1.0f / red[0];
  for (int i = tid; i < N; i += 1024) prob[i] = __expf(logits[i] - mx) * inv;
}

// ---------------------------------------------------------------------------
static void launch_gemm(hipStream_t s,
                        const unsigned short* A0, const int* I0, int S0, int B1,
                        const unsigned short* A1, const int* I1, int S1, int B2,
                        const unsigned short* A2, const int* I2, int S2,
                        const float* W, const float* bias,
                        int Kpad, int Kreal, int Nout,
                        float* outF, unsigned short* outB, int M, int act) {
  int mtiles = M >> 4;
  dim3 grid((mtiles + 7) / 8, Nout / 64);
  size_t shmem = (size_t)Kpad * 128;   // (K/32)*1024 dwords
  k_gemm_wmma<<<grid, 256, shmem, s>>>(A0, I0, S0, B1, A1, I1, S1, B2, A2, I2, S2,
                                       W, bias, Kpad, Kreal, Nout, outF, outB,
                                       mtiles, act);
}

extern "C" void kernel_launch(void* const* d_in, const int* in_sizes, int n_in,
                              void* d_out, int out_size, void* d_ws, size_t ws_size,
                              hipStream_t stream) {
  (void)in_sizes; (void)n_in; (void)out_size; (void)ws_size;
  const int N = NN, E = NE;

  const float* nodes     = (const float*)d_in[0];
  const float* edges     = (const float*)d_in[1];
  const int*   senders   = (const int*)d_in[2];
  const int*   receivers = (const int*)d_in[3];
  const float *faW1 = (const float*)d_in[4],  *fab1 = (const float*)d_in[5];
  const float *faW2 = (const float*)d_in[6],  *fab2 = (const float*)d_in[7];
  const float *fbW1 = (const float*)d_in[8],  *fbb1 = (const float*)d_in[9];
  const float *fbW2 = (const float*)d_in[10], *fbb2 = (const float*)d_in[11];
  const float *fvW1 = (const float*)d_in[12], *fvb1 = (const float*)d_in[13];
  const float *fvW2 = (const float*)d_in[14], *fvb2 = (const float*)d_in[15];
  const float *feW1 = (const float*)d_in[16], *feb1 = (const float*)d_in[17];
  const float *feW2 = (const float*)d_in[18], *feb2 = (const float*)d_in[19];
  const float *m1W1 = (const float*)d_in[20], *m1b1 = (const float*)d_in[21];
  const float *m1W2 = (const float*)d_in[22], *m1b2 = (const float*)d_in[23];
  const float *m2W1 = (const float*)d_in[24], *m2b1 = (const float*)d_in[25];
  const float *m2W2 = (const float*)d_in[26], *m2b2 = (const float*)d_in[27];

  // ---- workspace carve-out (deterministic) ----
  char* ws = (char*)d_ws;
  size_t off = 0;
  auto walloc = [&](size_t bytes) -> char* {
    char* p = ws + off;
    off += (bytes + 255) & ~(size_t)255;
    return p;
  };
  unsigned short* nodes_bf = (unsigned short*)walloc((size_t)N * 32 * 2);
  unsigned short* edges_bf = (unsigned short*)walloc((size_t)E * 32 * 2);
  unsigned short* n_bf     = (unsigned short*)walloc((size_t)N * 128 * 2);
  unsigned short* e_bf     = (unsigned short*)walloc((size_t)E * 64 * 2);
  unsigned short* hn_bf    = (unsigned short*)walloc((size_t)N * 128 * 2);
  unsigned short* he_bf    = (unsigned short*)walloc((size_t)E * 64 * 2);
  unsigned short* agg_bf   = (unsigned short*)walloc((size_t)N * 128 * 2);
  float* sum_s  = (float*)walloc((size_t)N * 64 * 4);
  float* sum_r  = (float*)walloc((size_t)N * 64 * 4);
  float* cnt_s  = (float*)walloc((size_t)N * 4);
  float* cnt_r  = (float*)walloc((size_t)N * 4);
  float* logits = (float*)walloc((size_t)N * 4);

  // ---- output slots: [prob N][disp 3N][n 128N][e 64E] ----
  float* prob = (float*)d_out;
  float* disp = prob + N;
  float* nF   = disp + (size_t)N * 3;
  float* eF   = nF + (size_t)N * 128;

  // ---- input conversion (f32 -> bf16, K padded to 32) ----
  k_convert_pad<<<(N * 32 + 255) / 256, 256, 0, stream>>>(nodes, nodes_bf, N, 8, 32);
  k_convert_pad<<<(E * 32 + 255) / 256, 256, 0, stream>>>(edges, edges_bf, E, 4, 32);

  // ---- fa: nodes [N,8] -> [N,128] -> [N,128] ----
  launch_gemm(stream, nodes_bf, nullptr, 32, 32, nodes_bf, nullptr, 32, 32,
              nodes_bf, nullptr, 32, faW1, fab1, 32, 8, 128, nullptr, hn_bf, N, 1);
  launch_gemm(stream, hn_bf, nullptr, 128, 128, hn_bf, nullptr, 128, 128,
              hn_bf, nullptr, 128, faW2, fab2, 128, 128, 128, nF, n_bf, N, 1);

  // ---- fb: edges [E,4] -> [E,64] -> [E,64] ----
  launch_gemm(stream, edges_bf, nullptr, 32, 32, edges_bf, nullptr, 32, 32,
              edges_bf, nullptr, 32, fbW1, fbb1, 32, 4, 64, nullptr, he_bf, E, 1);
  launch_gemm(stream, he_bf, nullptr, 64, 64, he_bf, nullptr, 64, 64,
              he_bf, nullptr, 64, fbW2, fbb2, 64, 64, 64, eF, e_bf, E, 1);

  // ---- 2 message-passing steps ----
  for (int step = 0; step < 2; ++step) {
    k_zero<<<(N * 128 + 255) / 256, 256, 0, stream>>>(sum_s, N * 128); // sum_s+sum_r contig
    k_zero<<<(N + 255) / 256, 256, 0, stream>>>(cnt_s, N);
    k_zero<<<(N + 255) / 256, 256, 0, stream>>>(cnt_r, N);
    k_scatter<<<(E * 16 + 255) / 256, 256, 0, stream>>>(eF, senders, receivers,
                                                        sum_s, sum_r, cnt_s, cnt_r, E);
    k_mean<<<(N * 128 + 255) / 256, 256, 0, stream>>>(sum_s, sum_r, cnt_s, cnt_r,
                                                      agg_bf, N);
    // fv: concat([n, agg_s, agg_r]) [N,256] -> [N,128] -> [N,128]
    launch_gemm(stream, n_bf, nullptr, 128, 128, agg_bf, nullptr, 128, 256,
                agg_bf, nullptr, 128, fvW1, fvb1, 256, 256, 128, nullptr, hn_bf, N, 1);
    launch_gemm(stream, hn_bf, nullptr, 128, 128, hn_bf, nullptr, 128, 128,
                hn_bf, nullptr, 128, fvW2, fvb2, 128, 128, 128, nF, n_bf, N, 1);
    // fe: concat([e, n[snd], n[rcv]]) [E,320] -> [E,64] -> [E,64] (fused gather)
    launch_gemm(stream, e_bf, nullptr, 64, 64, n_bf, senders, 128, 192,
                n_bf, receivers, 128, feW1, feb1, 320, 320, 64, nullptr, he_bf, E, 1);
    launch_gemm(stream, he_bf, nullptr, 64, 64, he_bf, nullptr, 64, 64,
                he_bf, nullptr, 64, feW2, feb2, 64, 64, 64, eF, e_bf, E, 1);
  }

  // ---- mlp1 head -> logits -> softmax(prob) ----
  launch_gemm(stream, n_bf, nullptr, 128, 128, n_bf, nullptr, 128, 128,
              n_bf, nullptr, 128, m1W1, m1b1, 128, 128, 128, nullptr, hn_bf, N, 1);
  k_head<<<(N + 255) / 256, 256, 0, stream>>>(hn_bf, m1W2, m1b2, logits, N, 1);
  k_softmax<<<1, 1024, 0, stream>>>(logits, prob, N);

  // ---- mlp2 head -> disp ----
  launch_gemm(stream, n_bf, nullptr, 128, 128, n_bf, nullptr, 128, 128,
              n_bf, nullptr, 128, m2W1, m2b1, 128, 128, 128, nullptr, hn_bf, N, 1);
  k_head<<<(N * 3 + 255) / 256, 256, 0, stream>>>(hn_bf, m2W2, m2b2, disp, N, 3);
}
